// MultiTurnDecoder_395136991650
// MI455X (gfx1250) — matrix operand order repaired and checked by reading, not compiled, not run on hardware
//
#include <hip/hip_runtime.h>
#include <hip/hip_bf16.h>

typedef __bf16 bf16_t;
typedef __bf16 v16bf __attribute__((ext_vector_type(16)));
typedef float  v8f   __attribute__((ext_vector_type(8)));

#define T_SEQ   2048
#define IN_DIM  1024
#define OUT_DIM 4096
#define MEM_DIM 256
#define M_MEM   8192
#define HGRU    128
#define HLSTM   256

__device__ __forceinline__ float sigmoidf_(float x){ return 1.f/(1.f+__expf(-x)); }

// ---------------- conversion kernels ----------------
__global__ void k_cvt_bf16(const float* __restrict__ s, bf16_t* __restrict__ d, int n){
  int i = blockIdx.x*blockDim.x + threadIdx.x;
  if(i < n) d[i] = (bf16_t)s[i];
}

// xbf[t,j] = bf16(c[t,j] + o[j]),  j = i & (IN_DIM-1)
__global__ void k_add_bcast_cvt(const float* __restrict__ c, const float* __restrict__ o,
                                bf16_t* __restrict__ d, int n){
  int i = blockIdx.x*blockDim.x + threadIdx.x;
  if(i < n) d[i] = (bf16_t)(c[i] + o[i & (IN_DIM-1)]);
}

// ---------------- WMMA GEMM ----------------
// C[M,N] = A[M,K](bf16, row-major) @ W[N,K]^T (bf16 row-major, PyTorch layout) + bias[N]
// grid.x = M/64, grid.y = N/256; 256 threads = 8 waves laid out 2(M) x 4(N).
// Each wave: 2 m-tiles x 4 n-tiles (32x64 output), 8 WMMAs per K-step with 4x A-reuse.
__global__ void k_gemm_bf16(const bf16_t* __restrict__ A, const bf16_t* __restrict__ W,
                            const float* __restrict__ bias, float* __restrict__ C,
                            int M, int N, int K){
  const int lane = threadIdx.x & 31;
  const int wave = threadIdx.x >> 5;
  const int mt = blockIdx.x * 64  + (wave >> 2) * 32;
  const int nt = blockIdx.y * 256 + (wave & 3) * 64;
  const int l15 = lane & 15;
  const int kg  = lane >> 4;
  // A fragment: lane l -> row M; K = kg*16 + i (contiguous 16 bf16)
  const bf16_t* arow0 = A + (size_t)(mt + l15)      * K + kg*16;
  const bf16_t* arow1 = A + (size_t)(mt + 16 + l15) * K + kg*16;
  // B fragment: lane l -> col N; K = kg*16 + i (contiguous 16 bf16)
  const bf16_t* wrow0 = W + (size_t)(nt + 0*16 + l15) * K + kg*16;
  const bf16_t* wrow1 = W + (size_t)(nt + 1*16 + l15) * K + kg*16;
  const bf16_t* wrow2 = W + (size_t)(nt + 2*16 + l15) * K + kg*16;
  const bf16_t* wrow3 = W + (size_t)(nt + 3*16 + l15) * K + kg*16;

  v8f acc00 = {}, acc01 = {}, acc02 = {}, acc03 = {};
  v8f acc10 = {}, acc11 = {}, acc12 = {}, acc13 = {};

  for(int k0 = 0; k0 < K; k0 += 32){
    // prefetch the streaming A operands one K-block ahead (speculative; OOB dropped)
    __builtin_prefetch(arow0 + k0 + 32, 0, 1);
    __builtin_prefetch(arow1 + k0 + 32, 0, 1);

    v16bf a0, a1, b0, b1, b2, b3;
#pragma unroll
    for(int i=0;i<8;i++){  a0[i] = arow0[k0 + i];     a1[i] = arow1[k0 + i]; }
#pragma unroll
    for(int i=8;i<16;i++){ a0[i] = arow0[k0 + i]; a1[i] = arow1[k0 + i]; }
#pragma unroll
    for(int i=0;i<16;i++){
      b0[i] = wrow0[k0 + i]; b1[i] = wrow1[k0 + i];
      b2[i] = wrow2[k0 + i]; b3[i] = wrow3[k0 + i];
    }
    acc00 = __builtin_amdgcn_wmma_f32_16x16x32_bf16(false, a0, false, b0, (short)0, acc00, false, false);
    acc01 = __builtin_amdgcn_wmma_f32_16x16x32_bf16(false, a0, false, b1, (short)0, acc01, false, false);
    acc02 = __builtin_amdgcn_wmma_f32_16x16x32_bf16(false, a0, false, b2, (short)0, acc02, false, false);
    acc03 = __builtin_amdgcn_wmma_f32_16x16x32_bf16(false, a0, false, b3, (short)0, acc03, false, false);
    acc10 = __builtin_amdgcn_wmma_f32_16x16x32_bf16(false, a1, false, b0, (short)0, acc10, false, false);
    acc11 = __builtin_amdgcn_wmma_f32_16x16x32_bf16(false, a1, false, b1, (short)0, acc11, false, false);
    acc12 = __builtin_amdgcn_wmma_f32_16x16x32_bf16(false, a1, false, b2, (short)0, acc12, false, false);
    acc13 = __builtin_amdgcn_wmma_f32_16x16x32_bf16(false, a1, false, b3, (short)0, acc13, false, false);
  }

  // C/D layout: element r -> row = tileM + kg*8 + r, col = tileN + (lane&15)
  const int ncol = l15;
#pragma unroll
  for(int r=0;r<8;r++){
    int m0 = mt + kg*8 + r;
    int m1 = m0 + 16;
    v8f* accs0[4] = { &acc00, &acc01, &acc02, &acc03 };
    v8f* accs1[4] = { &acc10, &acc11, &acc12, &acc13 };
#pragma unroll
    for(int j=0;j<4;j++){
      int n = nt + j*16 + ncol;
      float bv = bias ? bias[n] : 0.f;
      C[(size_t)m0*N + n] = (*accs0[j])[r] + bv;
      C[(size_t)m1*N + n] = (*accs1[j])[r] + bv;
    }
  }
}

// ---------------- BiGRU tail (only last row of hs is needed) ----------------
// xg: [T, 768] (both directions packed: [dir0 3H | dir1 3H]); Whh: [2,384,128]; bhh: [2,384]
// block 0: forward full scan -> u[0..127]; block 1: single backward step at t=T-1 -> u[128..255]
__global__ void k_bigru_tail(const float* __restrict__ xg, const float* __restrict__ Whh,
                             const float* __restrict__ bhh, float* __restrict__ u){
  const int tid = threadIdx.x;           // 384 threads
  const int dir = blockIdx.x;
  const float* Wd = Whh + (size_t)dir*3*HGRU*HGRU;
  const float* bd = bhh + dir*3*HGRU;
  __shared__ float h[HGRU];
  __shared__ float hg[3*HGRU];
  if(dir == 0){
    if(tid < HGRU) h[tid] = 0.f;
    __syncthreads();
    for(int t=0; t<T_SEQ; t++){
      const float* wr = Wd + (size_t)tid*HGRU;
      float acc = bd[tid];
#pragma unroll 4
      for(int j=0;j<HGRU;j++) acc += wr[j]*h[j];
      hg[tid] = acc;
      __syncthreads();
      if(tid < HGRU){
        const float* g = xg + (size_t)t*768;
        float r = sigmoidf_(g[tid]         + hg[tid]);
        float z = sigmoidf_(g[HGRU+tid]    + hg[HGRU+tid]);
        float n = tanhf   (g[2*HGRU+tid] + r*hg[2*HGRU+tid]);
        h[tid] = (1.f - z)*n + z*h[tid];
      }
      __syncthreads();
    }
    if(tid < HGRU) u[tid] = h[tid];
  } else {
    // reverse scan's hs[T-1] == one GRU step on x[T-1] from h=0  (h@Whh.T = 0)
    if(tid < HGRU){
      const float* g = xg + (size_t)(T_SEQ-1)*768 + 384;
      float r = sigmoidf_(g[tid]         + bd[tid]);
      float z = sigmoidf_(g[HGRU+tid]    + bd[HGRU+tid]);
      float n = tanhf   (g[2*HGRU+tid] + r*bd[2*HGRU+tid]);
      u[HGRU + tid] = (1.f - z)*n;
    }
  }
}

// ---------------- BiLSTM (full sequence), one direction per block ----------------
// xg: [T, 2048] ([dir0 4H | dir1 4H]); Whh: [2,1024,256]; bhh: [2,1024]; out: [T,512]
__global__ void k_bilstm(const float* __restrict__ xg, const float* __restrict__ Whh,
                         const float* __restrict__ bhh, float* __restrict__ out){
  const int tid = threadIdx.x;           // 1024 threads
  const int dir = blockIdx.x;
  const float* Wd = Whh + (size_t)dir*4*HLSTM*HLSTM;
  const float* bd = bhh + dir*4*HLSTM;
  __shared__ float h[HLSTM];
  __shared__ float cst[HLSTM];
  __shared__ float hg[4*HLSTM];
  if(tid < HLSTM){ h[tid]=0.f; cst[tid]=0.f; }
  __syncthreads();
  for(int s=0; s<T_SEQ; s++){
    int t = (dir==0) ? s : (T_SEQ-1-s);
    const float* wr = Wd + (size_t)tid*HLSTM;
    float acc = bd[tid] + xg[(size_t)t*2048 + dir*1024 + tid];
#pragma unroll 4
    for(int j=0;j<HLSTM;j++) acc += wr[j]*h[j];
    hg[tid] = acc;
    __syncthreads();
    if(tid < HLSTM){
      float ig = sigmoidf_(hg[tid]);
      float fg = sigmoidf_(hg[HLSTM+tid]);
      float gg = tanhf   (hg[2*HLSTM+tid]);
      float og = sigmoidf_(hg[3*HLSTM+tid]);
      float cv = fg*cst[tid] + ig*gg;
      cst[tid] = cv;
      float hv = og*tanhf(cv);
      h[tid] = hv;
      out[(size_t)t*512 + dir*HLSTM + tid] = hv;
    }
    __syncthreads();
  }
}

// ---------------- attention ----------------
__global__ void k_attn_scores(const float* __restrict__ mem, const float* __restrict__ u,
                              float* __restrict__ scores){
  int m = blockIdx.x*blockDim.x + threadIdx.x;
  if(m < M_MEM){
    const float* row = mem + (size_t)m*MEM_DIM;
    float acc = 0.f;
#pragma unroll 4
    for(int j=0;j<MEM_DIM;j++) acc += row[j]*u[j];
    scores[m] = acc;
  }
}

__global__ void k_softmax_vec(float* __restrict__ s, int n){   // 1 block, 1024 threads
  __shared__ float red[1024];
  int tid = threadIdx.x;
  float mx = -3.4e38f;
  for(int i=tid;i<n;i+=1024) mx = fmaxf(mx, s[i]);
  red[tid] = mx; __syncthreads();
  for(int st=512; st>0; st>>=1){ if(tid<st) red[tid]=fmaxf(red[tid],red[tid+st]); __syncthreads(); }
  mx = red[0]; __syncthreads();
  float sm = 0.f;
  for(int i=tid;i<n;i+=1024) sm += __expf(s[i]-mx);
  red[tid] = sm; __syncthreads();
  for(int st=512; st>0; st>>=1){ if(tid<st) red[tid]+=red[tid+st]; __syncthreads(); }
  float inv = 1.f/red[0];
  for(int i=tid;i<n;i+=1024) s[i] = __expf(s[i]-mx)*inv;
}

__global__ void k_attn_gather(const float* __restrict__ p, const float* __restrict__ mem,
                              float* __restrict__ hvec){
  int j = threadIdx.x;                    // 256 threads, coalesced column walk
  float acc = 0.f;
  for(int m=0;m<M_MEM;m++) acc += p[m]*mem[(size_t)m*MEM_DIM + j];
  hvec[j] = acc;
}

__global__ void k_know(const float* __restrict__ u, const float* __restrict__ hvec,
                       const float* __restrict__ W, const float* __restrict__ b,
                       float* __restrict__ o){
  int n = threadIdx.x;                    // 1024 threads, 1 block
  __shared__ float uh[MEM_DIM];
  if(n < MEM_DIM) uh[n] = u[n] + hvec[n];
  __syncthreads();
  const float* wr = W + (size_t)n*MEM_DIM;
  float acc = b[n];
#pragma unroll 4
  for(int j=0;j<MEM_DIM;j++) acc += wr[j]*uh[j];
  o[n] = acc;
}

// ---------------- row softmax (in place) ----------------
__global__ void k_row_softmax(float* __restrict__ x, int ncols){ // block per row, 256 thr
  __shared__ float red[256];
  int t = blockIdx.x, tid = threadIdx.x;
  float* row = x + (size_t)t*ncols;
  float mx = -3.4e38f;
  for(int i=tid;i<ncols;i+=256) mx = fmaxf(mx, row[i]);
  red[tid] = mx; __syncthreads();
  for(int st=128; st>0; st>>=1){ if(tid<st) red[tid]=fmaxf(red[tid],red[tid+st]); __syncthreads(); }
  mx = red[0]; __syncthreads();
  float sm = 0.f;
  for(int i=tid;i<ncols;i+=256) sm += __expf(row[i]-mx);
  red[tid] = sm; __syncthreads();
  for(int st=128; st>0; st>>=1){ if(tid<st) red[tid]+=red[tid+st]; __syncthreads(); }
  float inv = 1.f/red[0];
  for(int i=tid;i<ncols;i+=256) row[i] = __expf(row[i]-mx)*inv;
}

// ---------------- launch ----------------
extern "C" void kernel_launch(void* const* d_in, const int* in_sizes, int n_in,
                              void* d_out, int out_size, void* d_ws, size_t ws_size,
                              hipStream_t stream){
  (void)in_sizes; (void)n_in; (void)out_size; (void)ws_size;
  const float* c        = (const float*)d_in[0];
  const float* memory   = (const float*)d_in[1];
  const float* enc_Wih  = (const float*)d_in[2];
  const float* enc_Whh  = (const float*)d_in[3];
  const float* enc_bih  = (const float*)d_in[4];
  const float* enc_bhh  = (const float*)d_in[5];
  const float* cenc_Wih = (const float*)d_in[6];
  const float* cenc_Whh = (const float*)d_in[7];
  const float* cenc_bih = (const float*)d_in[8];
  const float* cenc_bhh = (const float*)d_in[9];
  const float* know_W   = (const float*)d_in[10];
  const float* know_b   = (const float*)d_in[11];
  const float* l0_Wih   = (const float*)d_in[12];
  const float* l0_Whh   = (const float*)d_in[13];
  const float* l0_bih   = (const float*)d_in[14];
  const float* l0_bhh   = (const float*)d_in[15];
  const float* l1_Wih   = (const float*)d_in[16];
  const float* l1_Whh   = (const float*)d_in[17];
  const float* l1_bih   = (const float*)d_in[18];
  const float* l1_bhh   = (const float*)d_in[19];
  const float* out_W    = (const float*)d_in[20];
  const float* out_b    = (const float*)d_in[21];

  float* out      = (float*)d_out;                       // [T*OUT] softmax result
  float* c_memory = out + (size_t)T_SEQ*OUT_DIM;         // tuple output #2: [256]

  // workspace layout
  char* ws = (char*)d_ws;
  size_t off = 0;
  auto take = [&](size_t bytes)->char*{
    char* p = ws + off; off = (off + bytes + 255) & ~(size_t)255; return p;
  };
  bf16_t* encW_bf   = (bf16_t*)take((size_t)2*3*HGRU*IN_DIM   * sizeof(bf16_t)); // 786432
  bf16_t* cencW_bf  = (bf16_t*)take((size_t)2*3*HGRU*IN_DIM   * sizeof(bf16_t));
  bf16_t* l0W_bf    = (bf16_t*)take((size_t)2*4*HLSTM*IN_DIM  * sizeof(bf16_t)); // 2097152
  bf16_t* l1W_bf    = (bf16_t*)take((size_t)2*4*HLSTM*512     * sizeof(bf16_t)); // 1048576
  bf16_t* outW_bf   = (bf16_t*)take((size_t)OUT_DIM*512       * sizeof(bf16_t)); // 2097152
  bf16_t* cbf       = (bf16_t*)take((size_t)T_SEQ*IN_DIM      * sizeof(bf16_t));
  bf16_t* xbf       = (bf16_t*)take((size_t)T_SEQ*IN_DIM      * sizeof(bf16_t));
  bf16_t* l0out_bf  = (bf16_t*)take((size_t)T_SEQ*512         * sizeof(bf16_t));
  bf16_t* l1out_bf  = (bf16_t*)take((size_t)T_SEQ*512         * sizeof(bf16_t));
  float*  xg        = (float*) take((size_t)T_SEQ*2048        * sizeof(float));  // shared by all proj GEMMs
  float*  l0out     = (float*) take((size_t)T_SEQ*512         * sizeof(float));
  float*  l1out     = (float*) take((size_t)T_SEQ*512         * sizeof(float));
  float*  uvec      = (float*) take(256  * sizeof(float));
  float*  scores    = (float*) take(M_MEM* sizeof(float));
  float*  hvec      = (float*) take(256  * sizeof(float));
  float*  ovec      = (float*) take(IN_DIM * sizeof(float));

  const int CT = 256;
  auto cvt = [&](const float* s, bf16_t* d, int n){
    k_cvt_bf16<<<(n + CT - 1)/CT, CT, 0, stream>>>(s, d, n);
  };

  // weight / activation conversion to bf16 (WMMA operand precision)
  cvt(enc_Wih,  encW_bf,  2*3*HGRU*IN_DIM);
  cvt(cenc_Wih, cencW_bf, 2*3*HGRU*IN_DIM);
  cvt(l0_Wih,   l0W_bf,   2*4*HLSTM*IN_DIM);
  cvt(l1_Wih,   l1W_bf,   2*4*HLSTM*512);
  cvt(out_W,    outW_bf,  OUT_DIM*512);
  cvt(c,        cbf,      T_SEQ*IN_DIM);

  // ---- encoders: xg = c @ Wih^T + bih (both directions in one WMMA GEMM, N=768) ----
  k_gemm_bf16<<<dim3(T_SEQ/64, 768/256), 256, 0, stream>>>(cbf, encW_bf,  enc_bih,  xg, T_SEQ, 768, IN_DIM);
  k_bigru_tail<<<2, 384, 0, stream>>>(xg, enc_Whh,  enc_bhh,  uvec);
  k_gemm_bf16<<<dim3(T_SEQ/64, 768/256), 256, 0, stream>>>(cbf, cencW_bf, cenc_bih, xg, T_SEQ, 768, IN_DIM);
  k_bigru_tail<<<2, 384, 0, stream>>>(xg, cenc_Whh, cenc_bhh, c_memory);

  // ---- memory attention + knowledge projection ----
  k_attn_scores<<<M_MEM/256, 256, 0, stream>>>(memory, uvec, scores);
  k_softmax_vec<<<1, 1024, 0, stream>>>(scores, M_MEM);
  k_attn_gather<<<1, 256, 0, stream>>>(scores, memory, hvec);
  k_know<<<1, 1024, 0, stream>>>(uvec, hvec, know_W, know_b, ovec);
  k_add_bcast_cvt<<<(T_SEQ*IN_DIM + CT - 1)/CT, CT, 0, stream>>>(c, ovec, xbf, T_SEQ*IN_DIM);

  // ---- BiLSTM layer 0 ----
  k_gemm_bf16<<<dim3(T_SEQ/64, 2048/256), 256, 0, stream>>>(xbf, l0W_bf, l0_bih, xg, T_SEQ, 2048, IN_DIM);
  k_bilstm<<<2, 1024, 0, stream>>>(xg, l0_Whh, l0_bhh, l0out);
  cvt(l0out, l0out_bf, T_SEQ*512);

  // ---- BiLSTM layer 1 ----
  k_gemm_bf16<<<dim3(T_SEQ/64, 2048/256), 256, 0, stream>>>(l0out_bf, l1W_bf, l1_bih, xg, T_SEQ, 2048, 512);
  k_bilstm<<<2, 1024, 0, stream>>>(xg, l1_Whh, l1_bhh, l1out);
  cvt(l1out, l1out_bf, T_SEQ*512);

  // ---- output projection (logits straight into d_out) + in-place row softmax ----
  k_gemm_bf16<<<dim3(T_SEQ/64, OUT_DIM/256), 256, 0, stream>>>(l1out_bf, outW_bf, out_b, out, T_SEQ, OUT_DIM, 512);
  k_row_softmax<<<T_SEQ, 256, 0, stream>>>(out, OUT_DIM);
}
